// GraphNetwork_77129022701723
// MI455X (gfx1250) — matrix operand order repaired and checked by reading, not compiled, not run on hardware
//
#include <hip/hip_runtime.h>
#include <hip/hip_bf16.h>

// ---------------------------------------------------------------------------
// Types for CDNA5 WMMA (wave32): D(16x16 f32) = A(16x32 bf16) x B(32x16 bf16) + C
// ---------------------------------------------------------------------------
typedef __attribute__((ext_vector_type(16))) __bf16 v16bf;
typedef __attribute__((ext_vector_type(8)))  __bf16 v8bf;
typedef __attribute__((ext_vector_type(8)))  float  v8f;

#if __has_builtin(__builtin_amdgcn_sched_barrier)
#define SCHED_FENCE() __builtin_amdgcn_sched_barrier(0)
#else
#define SCHED_FENCE()
#endif

// LDS tile strides (in bf16 elements). 40 halves = 80 bytes per row:
// keeps every fragment load 16-byte aligned and staggers banks.
#define A_STRIDE 40
#define B_STRIDE 40

// ---------------------------------------------------------------------------
// Fused GEMM kernel:
//   out[row, 0..255] = relu( concat_row @ W + bias )
// mode 0 (edge MLP, K=896): concat = [edges | nodes[senders] | nodes[receivers] | u]
//     epilogue also does agg[receivers[row]] += out (segment_sum) and col-sums.
// mode 1 (node MLP, K=640): concat = [agg | nodes | u], epilogue does col-sums.
// Block: 256 threads (8 waves) computes a 64-row x 256-col output tile.
// Wave w: rows 16*(w>>1)..+15, cols 128*(w&1)..+127  -> 8 WMMA C-tiles.
// Software pipeline: global loads for slab k+1 overlap WMMA compute of slab k
// via ping-pong LDS buffers (one barrier per K-step). A sched-barrier keeps
// the 18 fragment ds_loads ahead of the 8 back-to-back WMMAs (single wait).
// ---------------------------------------------------------------------------
__global__ __launch_bounds__(256) void gn_gemm_kernel(
    int mode, int rows, int K,
    const float* __restrict__ primary,   // edges (mode 0) or agg (mode 1), row len 256
    const float* __restrict__ nodes,     // [N,256]
    const int*   __restrict__ senders,
    const int*   __restrict__ receivers,
    const float* __restrict__ globals_,  // [128]
    const float* __restrict__ W,         // [K,256] row-major
    const float* __restrict__ bias,      // [256]
    float* __restrict__ outp,            // [rows,256]
    float* __restrict__ agg_out,         // [N,256] (mode 0 only)
    float* __restrict__ sum_out)         // [256]
{
    __shared__ __bf16 As[2][64 * A_STRIDE];    // ping-pong: 64 rows x 32 k
    __shared__ __bf16 Bt[2][256 * B_STRIDE];   // ping-pong: 256 n x 32 k (W transposed)
    __shared__ float  colsum[256];

    const int tid  = threadIdx.x;
    const int lane = tid & 31;
    const int wid  = tid >> 5;       // 0..7
    const int mt   = wid >> 1;       // 0..3: wave's 16-row group within 64-row tile
    const int nh   = wid & 1;        // 0..1: wave's 128-col half
    const long rowBase = (long)blockIdx.x * 64;

    colsum[tid] = 0.0f;

    // ---- staging geometry (hoisted; loop-invariant) ----
    const int sr  = tid >> 3;         // A row 0..31 (and +32)
    const int sc4 = (tid & 7) * 4;    // A col group 0..28
    const int wk  = tid >> 6;         // W k base 0..3 (+4 per it)
    const int wn4 = (tid & 63) * 4;   // W n group 0..252

    // Per-thread region base pointers (gather indices resolved ONCE).
    const float* aS[2][4];
    #pragma unroll
    for (int r = 0; r < 2; ++r) {
        long row = rowBase + sr + 32 * r;
        if (row >= rows) row = rows - 1;           // clamp (stores are guarded)
        if (mode == 0) {
            aS[r][0] = primary + row * 256;
            aS[r][1] = nodes + (long)senders[row]   * 256;
            aS[r][2] = nodes + (long)receivers[row] * 256;
            aS[r][3] = globals_;
        } else {
            aS[r][0] = primary + row * 256;
            aS[r][1] = nodes + row * 256;
            aS[r][2] = globals_;
            aS[r][3] = globals_;                   // never selected in mode 1
        }
    }

    v8f acc[8] = {};
    float4 av[2];      // in-flight A values for next slab
    float4 wv[8];      // in-flight W values for next slab

    // region = k0>>8 works for both modes (boundaries at 256/512/768).
    auto loadGlobal = [&](int kb) {
        const int k0     = kb << 5;
        const int region = k0 >> 8;
        const int off    = (k0 & 255) + sc4;
        #pragma unroll
        for (int r = 0; r < 2; ++r) {
            const float* s = (region == 0) ? aS[r][0]
                           : (region == 1) ? aS[r][1]
                           : (region == 2) ? aS[r][2] : aS[r][3];
            av[r] = *(const float4*)(s + off);
        }
        const float* wbase = W + (long)k0 * 256;
        #pragma unroll
        for (int it = 0; it < 8; ++it)
            wv[it] = *(const float4*)(wbase + (long)(wk + it * 4) * 256 + wn4);
    };

    auto storeLds = [&](int buf) {
        #pragma unroll
        for (int r = 0; r < 2; ++r) {
            __bf16* dst = &As[buf][(sr + 32 * r) * A_STRIDE + sc4];
            dst[0] = (__bf16)av[r].x; dst[1] = (__bf16)av[r].y;
            dst[2] = (__bf16)av[r].z; dst[3] = (__bf16)av[r].w;
        }
        #pragma unroll
        for (int it = 0; it < 8; ++it) {
            const int k = wk + it * 4;
            Bt[buf][(wn4 + 0) * B_STRIDE + k] = (__bf16)wv[it].x;
            Bt[buf][(wn4 + 1) * B_STRIDE + k] = (__bf16)wv[it].y;
            Bt[buf][(wn4 + 2) * B_STRIDE + k] = (__bf16)wv[it].z;
            Bt[buf][(wn4 + 3) * B_STRIDE + k] = (__bf16)wv[it].w;
        }
    };

    auto compute = [&](int buf) {
        // A fragment (16x32): lane l -> row (l&15); halves 0-7 = K[(l>>4)*8..],
        // halves 8-15 = same + 16.
        const __bf16* ap =
            &As[buf][(16 * mt + (lane & 15)) * A_STRIDE + ((lane >> 4) * 8)];
        const v8bf alo = *(const v8bf*)ap;
        const v8bf ahi = *(const v8bf*)(ap + 16);
        const v16bf a = __builtin_shufflevector(alo, ahi,
            0, 1, 2, 3, 4, 5, 6, 7, 8, 9, 10, 11, 12, 13, 14, 15);
        // Load ALL 8 B fragments first (kept live), then 8 WMMAs back-to-back.
        v16bf bfr[8];
        #pragma unroll
        for (int j = 0; j < 8; ++j) {
            const int n = nh * 128 + 16 * j + (lane & 15);
            const __bf16* bp = &Bt[buf][n * B_STRIDE + ((lane >> 4) * 8)];
            const v8bf blo = *(const v8bf*)bp;
            const v8bf bhi = *(const v8bf*)(bp + 16);
            bfr[j] = __builtin_shufflevector(blo, bhi,
                0, 1, 2, 3, 4, 5, 6, 7, 8, 9, 10, 11, 12, 13, 14, 15);
        }
        SCHED_FENCE();   // keep ds_loads ahead of the WMMA burst (one dscnt wait)
        #pragma unroll
        for (int j = 0; j < 8; ++j)
            acc[j] = __builtin_amdgcn_wmma_f32_16x16x32_bf16(
                false, a, false, bfr[j], (short)0, acc[j], false, false);
    };

    // ---- software-pipelined main loop (1 barrier per K-step) ----
    const int nK = K >> 5;
    loadGlobal(0);
    storeLds(0);
    __syncthreads();
    for (int kb = 0; kb < nK; ++kb) {
        const int cur  = kb & 1;
        const bool more = (kb + 1) < nK;
        if (more) loadGlobal(kb + 1);                 // overlap with compute(cur)
        if (kb + 2 < nK)                              // prefetch W two slabs ahead
            __builtin_prefetch(W + ((long)(kb + 2) << 5) * 256 + tid * 32, 0, 1);
        compute(cur);
        if (more) storeLds(cur ^ 1);
        __syncthreads();
    }

    // ---- epilogue: bias + relu + store; fused scatter + column sums ----
    const int g   = lane >> 4;   // C layout: elt i of VGPR -> row i + 8*g
    const int nlo = lane & 15;

    // Hoist receiver gathers: rows are identical across the 8 column tiles.
    long rowIdx[8];
    long rcvOff[8];
    #pragma unroll
    for (int i = 0; i < 8; ++i) {
        const long row = rowBase + 16 * mt + i + 8 * g;
        rowIdx[i] = row;
        rcvOff[i] = (mode == 0 && row < rows) ? (long)receivers[row] * 256 : 0;
    }

    #pragma unroll
    for (int j = 0; j < 8; ++j) {
        const int col   = nh * 128 + 16 * j + nlo;
        const float bv  = bias[col];
        float csum = 0.0f;
        #pragma unroll
        for (int i = 0; i < 8; ++i) {
            const long row = rowIdx[i];
            float v = acc[j][i] + bv;
            v = v > 0.0f ? v : 0.0f;
            if (row < rows) {
                outp[row * 256 + col] = v;
                if (mode == 0)
                    atomicAdd(agg_out + rcvOff[i] + col, v);  // segment_sum
                csum += v;
            }
        }
        atomicAdd(&colsum[col], csum);   // ds_add_f32
    }
    __syncthreads();
    atomicAdd(sum_out + tid, colsum[tid]);
}

// ---------------------------------------------------------------------------
// Zero-init workspace (agg + sum_e + sum_n) — must run every launch.
// ---------------------------------------------------------------------------
__global__ void gn_zero_kernel(float* __restrict__ p, long n) {
    long i = (long)blockIdx.x * blockDim.x + threadIdx.x;
    const long stride = (long)gridDim.x * blockDim.x;
    for (; i < n; i += stride) p[i] = 0.0f;
}

// ---------------------------------------------------------------------------
// Global MLP: new_globals = relu( [sum_e | sum_n | u] @ Wg + bg ), 640x128
// ---------------------------------------------------------------------------
__global__ __launch_bounds__(128) void gn_global_kernel(
    const float* __restrict__ sum_e, const float* __restrict__ sum_n,
    const float* __restrict__ globals_, const float* __restrict__ Wg,
    const float* __restrict__ bg, float* __restrict__ outp)
{
    const int n = threadIdx.x;          // 0..127
    float acc = bg[n];
    for (int k = 0; k < 256; ++k) acc += sum_e[k] * Wg[(long)k * 128 + n];
    for (int k = 0; k < 256; ++k) acc += sum_n[k] * Wg[(long)(256 + k) * 128 + n];
    for (int k = 0; k < 128; ++k) acc += globals_[k] * Wg[(long)(512 + k) * 128 + n];
    outp[n] = acc > 0.0f ? acc : 0.0f;
}

// ---------------------------------------------------------------------------
extern "C" void kernel_launch(void* const* d_in, const int* in_sizes, int n_in,
                              void* d_out, int out_size, void* d_ws, size_t ws_size,
                              hipStream_t stream) {
    (void)n_in; (void)out_size; (void)ws_size;
    const float* nodes     = (const float*)d_in[0];
    const float* edges     = (const float*)d_in[1];
    const float* globals_  = (const float*)d_in[2];
    const int*   senders   = (const int*)  d_in[3];
    const int*   receivers = (const int*)  d_in[4];
    const float* We        = (const float*)d_in[5];
    const float* be        = (const float*)d_in[6];
    const float* Wn        = (const float*)d_in[7];
    const float* bn        = (const float*)d_in[8];
    const float* Wg        = (const float*)d_in[9];
    const float* bg        = (const float*)d_in[10];

    const int E = in_sizes[3];           // 200000 edges
    const int N = in_sizes[0] / 256;     // 50000 nodes

    float* out_edges = (float*)d_out;                       // [E,256]
    float* out_nodes = out_edges + (long)E * 256;           // [N,256]
    float* out_glob  = out_nodes + (long)N * 256;           // [128]

    float* agg   = (float*)d_ws;                            // [N,256]
    float* sum_e = agg + (long)N * 256;                     // [256]
    float* sum_n = sum_e + 256;                             // [256]

    const long zeroN = (long)N * 256 + 512;
    gn_zero_kernel<<<1024, 256, 0, stream>>>(agg, zeroN);

    // Edge MLP (K = 256+256+256+128 = 896) + fused segment_sum + col-sums
    gn_gemm_kernel<<<(E + 63) / 64, 256, 0, stream>>>(
        0, E, 896, edges, nodes, senders, receivers, globals_,
        We, be, out_edges, agg, sum_e);

    // Node MLP (K = 256+256+128 = 640) + col-sums
    gn_gemm_kernel<<<(N + 63) / 64, 256, 0, stream>>>(
        1, N, 640, agg, nodes, nullptr, nullptr, globals_,
        Wn, bn, out_nodes, nullptr, sum_n);

    // Global MLP
    gn_global_kernel<<<1, 128, 0, stream>>>(sum_e, sum_n, globals_, Wg, bg, out_glob);
}
